// LightGCN_128849019430
// MI455X (gfx1250) — matrix operand order repaired and checked by reading, not compiled, not run on hardware
//
#include <hip/hip_runtime.h>
#include <hip/hip_bf16.h>
#include <math.h>

// Problem constants (from reference)
#define U_CNT   100000
#define I_CNT   50000
#define N_CNT   150000
#define E_DIM   64
#define B_CNT   3
#define NNZ_CNT 2400000
#define M_ROWS  (N_CNT * B_CNT)   // 450000, divisible by 16 -> 28125 tiles

typedef __attribute__((ext_vector_type(16))) __bf16 v16bf;
typedef __attribute__((ext_vector_type(8)))  float  v8f;

// ---------------------------------------------------------------- utilities
__global__ __launch_bounds__(256) void zero_f4(float4* __restrict__ p) {
  size_t i = (size_t)blockIdx.x * blockDim.x + threadIdx.x;
  p[i] = make_float4(0.f, 0.f, 0.f, 0.f);
}

// ego = concat(user_emb, item_emb); acc = ego; cur = ego
__global__ __launch_bounds__(256) void init_ego(const float* __restrict__ ue,
                                                const float* __restrict__ ie,
                                                float* __restrict__ acc,
                                                float* __restrict__ cur) {
  size_t gid = (size_t)blockIdx.x * blockDim.x + threadIdx.x;  // N*16 threads
  size_t n = gid >> 4;
  int q = (int)(gid & 15);
  const float4 v = (n < U_CNT)
      ? *(const float4*)(ue + n * E_DIM + (size_t)q * 4)
      : *(const float4*)(ie + (n - U_CNT) * E_DIM + (size_t)q * 4);
  *(float4*)(acc + n * E_DIM + (size_t)q * 4) = v;
  *(float4*)(cur + n * E_DIM + (size_t)q * 4) = v;
}

// y[rows[k], :] += vals[k] * x[cols[k], :] ; 16 lanes per edge, float4 each
__global__ __launch_bounds__(256) void spmm_scatter(const float* __restrict__ vals,
                                                    const int*   __restrict__ rows,
                                                    const int*   __restrict__ cols,
                                                    const float* __restrict__ x,
                                                    float*       __restrict__ y) {
  size_t gid = (size_t)blockIdx.x * blockDim.x + threadIdx.x;  // NNZ*16 threads
  size_t k = gid >> 4;
  int q = (int)(gid & 15);
  float v = vals[k];
  int r = rows[k];
  int c = cols[k];
  const float4 xv = *(const float4*)(x + (size_t)c * E_DIM + (size_t)q * 4);
  float* yp = y + (size_t)r * E_DIM + (size_t)q * 4;
  atomicAdd(yp + 0, v * xv.x);
  atomicAdd(yp + 1, v * xv.y);
  atomicAdd(yp + 2, v * xv.z);
  atomicAdd(yp + 3, v * xv.w);
}

__global__ __launch_bounds__(256) void acc_add(float4* __restrict__ acc,
                                               const float4* __restrict__ src) {
  size_t i = (size_t)blockIdx.x * blockDim.x + threadIdx.x;
  float4 a = acc[i];
  const float4 s = src[i];
  a.x += s.x; a.y += s.y; a.z += s.z; a.w += s.w;
  acc[i] = a;
}

// --------------------------------------------------- WMMA attention scores
// score[j, m] = tanh( (beh_row(m)/4) @ W1[j] ) @ W2[j],  m = n*3 + b
// One wave (32 threads) per 16-row tile; K=64 via two 16x16x32 bf16 WMMAs.
__global__ __launch_bounds__(32) void score_wmma(const float* __restrict__ beh0,
                                                 const float* __restrict__ beh1,
                                                 const float* __restrict__ beh2,
                                                 const float* __restrict__ W1,
                                                 const float* __restrict__ W2,
                                                 float* __restrict__ score) {
  const int tile  = blockIdx.x;        // 0..28124
  const int j     = blockIdx.y;        // weight-set 0..2
  const int lane  = threadIdx.x;       // 0..31 (wave32)
  const int n16   = lane & 15;
  const int khalf = lane >> 4;

  // ---- B fragments: W1[j] is 64x16 row-major (e, c).
  // B 32x16 as loaded here: lane = column n16, halves of wave split K by 16.
  const float* __restrict__ w1 = W1 + (size_t)j * (E_DIM * 16);
  v16bf bf0, bf1;
#pragma unroll
  for (int i = 0; i < 16; ++i) {
    const int k0 = khalf * 16 + i;
    bf0[i] = (__bf16)w1[k0 * 16 + n16];          // K = 0..31
    bf1[i] = (__bf16)w1[(k0 + 32) * 16 + n16];   // K = 32..63
  }

  // ---- A fragment: row m of the flattened (n,b) matrix.
  // 16-bit A 16x32 layout: lane -> row M = lane&15; this code splits K by 16 per half, as for B.
  const int m  = tile * 16 + n16;
  const int bb = m % B_CNT;
  const int nn = m / B_CNT;
  const float* __restrict__ arow =
      (bb == 0 ? beh0 : (bb == 1 ? beh1 : beh2)) + (size_t)nn * E_DIM;

  v16bf af0, af1;
#pragma unroll
  for (int i = 0; i < 8; ++i) {
    af0[i]     = (__bf16)(0.25f * arow[khalf * 16 + i]);       // K = 16*khalf + 0..7
    af0[i + 8] = (__bf16)(0.25f * arow[khalf * 16 + 8 + i]);   // K = 16*khalf + 8..15
    af1[i]     = (__bf16)(0.25f * arow[32 + khalf * 16 + i]);  // K = 32 + 16*khalf + 0..7
    af1[i + 8] = (__bf16)(0.25f * arow[40 + khalf * 16 + i]);  // K = 32 + 16*khalf + 8..15
  }

  v8f c = {};
  c = __builtin_amdgcn_wmma_f32_16x16x32_bf16(false, af0, false, bf0,
                                              (short)0, c, false, false);
  c = __builtin_amdgcn_wmma_f32_16x16x32_bf16(false, af1, false, bf1,
                                              (short)0, c, false, false);

  // C/D layout: lane owns column N=n16; reg r holds row M = r + 8*khalf.
  // score[M] = sum_c tanh(c[M][c]) * w2[c]  -> per-lane scale + 16-lane reduce
  const float w2v = W2[j * 16 + n16];
  float s[8];
#pragma unroll
  for (int r = 0; r < 8; ++r) s[r] = tanhf(c[r]) * w2v;
#pragma unroll
  for (int off = 1; off < 16; off <<= 1) {
#pragma unroll
    for (int r = 0; r < 8; ++r) s[r] += __shfl_xor(s[r], off, 32);
  }
  if (n16 == 0) {
#pragma unroll
    for (int r = 0; r < 8; ++r)
      score[(size_t)j * M_ROWS + (size_t)tile * 16 + khalf * 8 + r] = s[r];
  }
}

// --------------------------------------------- softmax over B + combine out
__global__ __launch_bounds__(256) void attn_out(const float* __restrict__ beh0,
                                                const float* __restrict__ beh1,
                                                const float* __restrict__ beh2,
                                                const float* __restrict__ score,
                                                float* __restrict__ out) {
  size_t gid = (size_t)blockIdx.x * blockDim.x + threadIdx.x;  // 3*N*16 threads
  const int q = (int)(gid & 15);
  size_t rest = gid >> 4;
  size_t n = rest % N_CNT;
  int j = (int)(rest / N_CNT);

  const float* sc = score + (size_t)j * M_ROWS + n * B_CNT;
  const float s0 = sc[0], s1 = sc[1], s2 = sc[2];
  const float mx = fmaxf(s0, fmaxf(s1, s2));
  const float e0 = __expf(s0 - mx), e1 = __expf(s1 - mx), e2 = __expf(s2 - mx);
  const float inv = 0.25f / (e0 + e1 + e2);  // fold beh = acc/4 scaling
  const float a0 = e0 * inv, a1 = e1 * inv, a2 = e2 * inv;

  const size_t eo = n * E_DIM + (size_t)q * 4;
  const float4 v0 = *(const float4*)(beh0 + eo);
  const float4 v1 = *(const float4*)(beh1 + eo);
  const float4 v2 = *(const float4*)(beh2 + eo);
  float4 o;
  o.x = a0 * v0.x + a1 * v1.x + a2 * v2.x;
  o.y = a0 * v0.y + a1 * v1.y + a2 * v2.y;
  o.z = a0 * v0.z + a1 * v1.z + a2 * v2.z;
  o.w = a0 * v0.w + a1 * v1.w + a2 * v2.w;

  size_t off;
  if (n < U_CNT)
    off = (size_t)j * U_CNT * E_DIM + n * E_DIM + (size_t)q * 4;
  else
    off = (size_t)B_CNT * U_CNT * E_DIM + (size_t)j * I_CNT * E_DIM +
          (n - U_CNT) * E_DIM + (size_t)q * 4;
  *(float4*)(out + off) = o;
}

// --------------------------------------------------------------- dispatcher
extern "C" void kernel_launch(void* const* d_in, const int* in_sizes, int n_in,
                              void* d_out, int out_size, void* d_ws, size_t ws_size,
                              hipStream_t stream) {
  (void)in_sizes; (void)n_in; (void)out_size; (void)ws_size;

  const float* user_emb = (const float*)d_in[0];
  const float* item_emb = (const float*)d_in[1];
  const float* W1       = (const float*)d_in[2];
  const float* W2       = (const float*)d_in[3];
  const float* adj_vals = (const float*)d_in[4];
  const int*   adj_rows = (const int*)d_in[5];
  const int*   adj_cols = (const int*)d_in[6];
  float* out = (float*)d_out;
  float* ws  = (float*)d_ws;

  const size_t NE = (size_t)N_CNT * E_DIM;  // 9.6M floats
  float* beh[B_CNT] = { ws, ws + NE, ws + 2 * NE };
  float* bufA  = ws + 3 * NE;
  float* bufB  = ws + 4 * NE;
  float* score = ws + 5 * NE;               // 3 * 450000 floats

  const int TPB = 256;
  const int blocks_NE4  = (int)(NE / 4 / TPB);               // 9375
  const int blocks_init = (int)((size_t)N_CNT * 16 / TPB);   // 9375
  const int blocks_spmm = (int)((size_t)NNZ_CNT * 16 / TPB); // 150000
  const int blocks_attn = (int)((size_t)B_CNT * N_CNT * 16 / TPB); // 28125

  for (int b = 0; b < B_CNT; ++b) {
    init_ego<<<blocks_init, TPB, 0, stream>>>(user_emb, item_emb, beh[b], bufA);
    float* cur = bufA;
    float* nxt = bufB;
    const float* vals = adj_vals + (size_t)b * NNZ_CNT;
    const int*   rows = adj_rows + (size_t)b * NNZ_CNT;
    const int*   cols = adj_cols + (size_t)b * NNZ_CNT;
    for (int l = 0; l < 3; ++l) {
      zero_f4<<<blocks_NE4, TPB, 0, stream>>>((float4*)nxt);
      spmm_scatter<<<blocks_spmm, TPB, 0, stream>>>(vals, rows, cols, cur, nxt);
      acc_add<<<blocks_NE4, TPB, 0, stream>>>((float4*)beh[b], (const float4*)nxt);
      float* t = cur; cur = nxt; nxt = t;
    }
  }

  score_wmma<<<dim3(M_ROWS / 16, B_CNT), 32, 0, stream>>>(
      beh[0], beh[1], beh[2], W1, W2, score);

  attn_out<<<blocks_attn, TPB, 0, stream>>>(beh[0], beh[1], beh[2], score, out);
}
